// EGCN_CPU_84121229460221
// MI455X (gfx1250) — matrix operand order repaired and checked by reading, not compiled, not run on hardware
//
#include <hip/hip_runtime.h>

#define N_NODES 100000
#define N_EDGES 600000
#define IN_F 128
#define HID_F 128
#define OUT_F 64

typedef __attribute__((ext_vector_type(16))) __bf16 v16bf;
typedef __attribute__((ext_vector_type(8)))  __bf16 v8bf;
typedef __attribute__((ext_vector_type(8)))  float  v8f;

// ---------------------------------------------------------------- utilities

__global__ void zero_kernel(float* __restrict__ p, size_t n4) {
  size_t i = (size_t)blockIdx.x * blockDim.x + threadIdx.x;
  if (i < n4) reinterpret_cast<float4*>(p)[i] = make_float4(0.f, 0.f, 0.f, 0.f);
}

__global__ void cvt_bf16_kernel(const float* __restrict__ src, __bf16* __restrict__ dst, int n) {
  int i = blockIdx.x * blockDim.x + threadIdx.x;
  if (i < n) dst[i] = (__bf16)src[i];
}

// degree: atomicAdd 1 per edge at dst
__global__ void deg_kernel(const int* __restrict__ dst, float* __restrict__ deg, int ne) {
  int i = blockIdx.x * blockDim.x + threadIdx.x;
  if (i < ne) atomicAdd(&deg[dst[i]], 1.0f);
}

__global__ void dinv_kernel(const float* __restrict__ deg, float* __restrict__ dinv, int n) {
  int i = blockIdx.x * blockDim.x + threadIdx.x;
  if (i < n) dinv[i] = rsqrtf(fmaxf(deg[i], 1.0f));
}

// out = -f   (vectorized; initializes T1 so scatter accumulates X1 directly)
__global__ void neg_kernel(const float* __restrict__ f, float* __restrict__ out) {
  size_t i = (size_t)blockIdx.x * blockDim.x + threadIdx.x;
  if (i >= (size_t)N_NODES * 32) return;
  float4 v = reinterpret_cast<const float4*>(f)[i];
  reinterpret_cast<float4*>(out)[i] = make_float4(-v.x, -v.y, -v.z, -v.w);
}

// out = -2*x1 - x0   (initializes T2 so scatter accumulates X2 directly)
__global__ void init2_kernel(const float* __restrict__ x1, const float* __restrict__ x0,
                             float* __restrict__ out) {
  size_t i = (size_t)blockIdx.x * blockDim.x + threadIdx.x;
  if (i >= (size_t)N_NODES * 32) return;
  float4 a = reinterpret_cast<const float4*>(x1)[i];
  float4 b = reinterpret_cast<const float4*>(x0)[i];
  reinterpret_cast<float4*>(out)[i] =
      make_float4(-2.f * a.x - b.x, -2.f * a.y - b.y, -2.f * a.z - b.z, -2.f * a.w - b.w);
}

// agg[dst] += x[src] * (coef * dinv[src] * dinv[dst])
// One wave per edge (32 lanes x 4 floats = 128 features). Edge metadata is
// wave-uniform: force it scalar with readfirstlane so it uses the SMEM path
// and the weight lives in an SGPR.
__global__ void scatter_kernel(const float* __restrict__ x, const float* __restrict__ dinv,
                               const int* __restrict__ src, const int* __restrict__ dst,
                               float* __restrict__ agg, float coef) {
  long long t = (long long)blockIdx.x * blockDim.x + threadIdx.x;
  int wave = (int)(t >> 5);
  if (wave >= N_EDGES) return;
  int e = __builtin_amdgcn_readfirstlane(wave);
  int s = __builtin_amdgcn_readfirstlane(src[e]);
  int d = __builtin_amdgcn_readfirstlane(dst[e]);
  float w = coef * dinv[s] * dinv[d];
  int c4 = (int)(t & 31) * 4;
  float4 v = *reinterpret_cast<const float4*>(x + (size_t)s * 128 + c4);
  float* ap = agg + (size_t)d * 128 + c4;
  atomicAdd(ap + 0, v.x * w);
  atomicAdd(ap + 1, v.y * w);
  atomicAdd(ap + 2, v.z * w);
  atomicAdd(ap + 3, v.w * w);
}

// ---------------------------------------------------------------- WMMA GEMM
// out[m, 0..NO) = epilogue( sum over NUMA inputs of A_i[m,:128] @ W[:, i*128..]^T + bias )
// W is bf16, row-major [NO x K], K = NUMA*128.
// Block: 16 node-rows, NO/16 waves; each wave owns one 16x16 output tile.
// MODE: 0 = relu+batchnorm, 1 = relu+residual, 2 = relu, 3 = none.
template <int NUMA, int NO, int MODE>
__global__ __launch_bounds__(NO * 2)
void gemm_kernel(const float* __restrict__ A0, const float* __restrict__ A1,
                 const float* __restrict__ A2, const __bf16* __restrict__ W,
                 const float* __restrict__ bias,
                 const float* __restrict__ bn_gamma, const float* __restrict__ bn_beta,
                 const float* __restrict__ bn_mean, const float* __restrict__ bn_var,
                 const float* __restrict__ resid, float* __restrict__ out) {
  constexpr int K = NUMA * 128;
  __shared__ __bf16 aLds[16 * K];

  const int    tid     = threadIdx.x;
  const size_t rowBase = (size_t)blockIdx.x * 16;

  // Stage the 16 x K activation tile into LDS, converting f32 -> bf16.
  for (int i = tid; i < 16 * K; i += NO * 2) {
    int r = i / K, k = i - r * K;
    const float* s = (NUMA == 1) ? A0 : (k < 128 ? A0 : (k < 256 ? A1 : A2));
    aLds[r * K + k] = (__bf16)s[(rowBase + r) * 128 + (k & 127)];
  }
  __syncthreads();

  const int lane  = tid & 31;
  const int ln    = lane & 15;
  const int hh    = lane >> 4;          // which 16-lane half
  const int ntile = tid >> 5;           // output column tile (wave id)

  v8f acc = {};
#pragma unroll
  for (int ks = 0; ks < K; ks += 32) {
    // A fragment: row M=ln; 16 contiguous K starting at ks + hh*16
    const __bf16* ar = &aLds[ln * K + ks + hh * 16];
    v8bf alo = *reinterpret_cast<const v8bf*>(ar);
    v8bf ahi = *reinterpret_cast<const v8bf*>(ar + 8);
    v16bf a = __builtin_shufflevector(alo, ahi, 0, 1, 2, 3, 4, 5, 6, 7,
                                      8, 9, 10, 11, 12, 13, 14, 15);
    // B fragment: col N = ntile*16+ln; 16 contiguous K starting at ks + hh*16
    v16bf b = *reinterpret_cast<const v16bf*>(&W[(size_t)(ntile * 16 + ln) * K + ks + hh * 16]);
    acc = __builtin_amdgcn_wmma_f32_16x16x32_bf16(false, a, false, b,
                                                  (short)0, acc, false, false);
  }

  // Epilogue: D VGPR r holds M = r + 8*hh, N = ntile*16 + ln
  const int col = ntile * 16 + ln;
  const float bv = bias[col];
  float bn_s = 0.f, bn_o = 0.f;
  if (MODE == 0) {
    bn_s = bn_gamma[col] * rsqrtf(bn_var[col] + 1e-5f);
    bn_o = bn_beta[col] - bn_mean[col] * bn_s;
  }
#pragma unroll
  for (int r = 0; r < 8; ++r) {
    size_t row = rowBase + r + 8 * hh;
    float v = acc[r] + bv;
    if (MODE <= 2) v = fmaxf(v, 0.0f);
    if (MODE == 0) v = v * bn_s + bn_o;
    if (MODE == 1) v += resid[row * 128 + col];
    out[row * NO + col] = v;
  }
}

// ---------------------------------------------------------------- launcher

extern "C" void kernel_launch(void* const* d_in, const int* in_sizes, int n_in,
                              void* d_out, int out_size, void* d_ws, size_t ws_size,
                              hipStream_t stream) {
  const float* features = (const float*)d_in[0];
  const int*   esrc     = (const int*)d_in[1];
  const int*   edst     = (const int*)d_in[2];
  const float* W1       = (const float*)d_in[3];
  const float* b1       = (const float*)d_in[4];
  const float* bn_gamma = (const float*)d_in[5];
  const float* bn_beta  = (const float*)d_in[6];
  const float* bn_mean  = (const float*)d_in[7];
  const float* bn_var   = (const float*)d_in[8];
  const float* W3       = (const float*)d_in[9];
  const float* b3       = (const float*)d_in[10];
  const float* Wm1      = (const float*)d_in[11];
  const float* bm1      = (const float*)d_in[12];
  const float* Wm2      = (const float*)d_in[13];
  const float* bm2      = (const float*)d_in[14];
  float* out = (float*)d_out;

  // workspace carve-up
  char* ws = (char*)d_ws;
  size_t off = 0;
  auto carve = [&](size_t bytes) { char* p = ws + off; off += (bytes + 255) & ~(size_t)255; return p; };
  const size_t NF = (size_t)N_NODES * 128;
  float* deg  = (float*)carve((size_t)N_NODES * 4);
  float* dinv = (float*)carve((size_t)N_NODES * 4);
  float* T1   = (float*)carve(NF * 4);
  float* T2   = (float*)carve(NF * 4);
  float* Y    = (float*)carve(NF * 4);
  float* Z    = (float*)carve(NF * 4);
  __bf16* Wb1  = (__bf16*)carve((size_t)128 * 384 * 2);
  __bf16* Wb3  = (__bf16*)carve((size_t)128 * 384 * 2);
  __bf16* Wbm1 = (__bf16*)carve((size_t)128 * 128 * 2);
  __bf16* Wbm2 = (__bf16*)carve((size_t)64 * 128 * 2);
  float* H = T1;  // reuse after conv2

  const int TPB = 256;
  const int gN    = (N_NODES + TPB - 1) / TPB;
  const int gE    = (N_EDGES + TPB - 1) / TPB;
  const int gScat = (int)(((long long)N_EDGES * 32 + TPB - 1) / TPB);
  const int gNF4  = (int)((NF / 4 + TPB - 1) / TPB);
  const int gMT   = N_NODES / 16;  // 6250

  // weights -> bf16 (L2-resident, reused by every block)
  cvt_bf16_kernel<<<(128 * 384 + TPB - 1) / TPB, TPB, 0, stream>>>(W1, Wb1, 128 * 384);
  cvt_bf16_kernel<<<(128 * 384 + TPB - 1) / TPB, TPB, 0, stream>>>(W3, Wb3, 128 * 384);
  cvt_bf16_kernel<<<(128 * 128 + TPB - 1) / TPB, TPB, 0, stream>>>(Wm1, Wbm1, 128 * 128);
  cvt_bf16_kernel<<<(64 * 128 + TPB - 1) / TPB, TPB, 0, stream>>>(Wm2, Wbm2, 64 * 128);

  // degrees and D^-1/2
  zero_kernel<<<(N_NODES / 4 + TPB - 1) / TPB, TPB, 0, stream>>>(deg, N_NODES / 4);
  deg_kernel<<<gE, TPB, 0, stream>>>(edst, deg, N_EDGES);
  dinv_kernel<<<gN, TPB, 0, stream>>>(deg, dinv, N_NODES);

  // ---- ChebConv 1 on X0 = features ----
  // X1 = agg(X0)*dinv - X0 : init T1 = -X0, scatter-accumulate with coef 1
  neg_kernel<<<gNF4, TPB, 0, stream>>>(features, T1);
  scatter_kernel<<<gScat, TPB, 0, stream>>>(features, dinv, esrc, edst, T1, 1.0f);
  // X2 = 2*agg(X1)*dinv - 2*X1 - X0 : init T2 = -2*X1 - X0, scatter coef 2
  init2_kernel<<<gNF4, TPB, 0, stream>>>(T1, features, T2);
  scatter_kernel<<<gScat, TPB, 0, stream>>>(T1, dinv, esrc, edst, T2, 2.0f);
  // Y = BN(relu([X0|X1|X2] @ W1^T + b1))
  gemm_kernel<3, 128, 0><<<gMT, 256, 0, stream>>>(features, T1, T2, Wb1, b1,
                                                  bn_gamma, bn_beta, bn_mean, bn_var,
                                                  nullptr, Y);

  // ---- ChebConv 2 on Y ----
  neg_kernel<<<gNF4, TPB, 0, stream>>>(Y, T1);
  scatter_kernel<<<gScat, TPB, 0, stream>>>(Y, dinv, esrc, edst, T1, 1.0f);
  init2_kernel<<<gNF4, TPB, 0, stream>>>(T1, Y, T2);
  scatter_kernel<<<gScat, TPB, 0, stream>>>(T1, dinv, esrc, edst, T2, 2.0f);
  // Z = relu([Y|X1'|X2'] @ W3^T + b3) + Y
  gemm_kernel<3, 128, 1><<<gMT, 256, 0, stream>>>(Y, T1, T2, Wb3, b3,
                                                  nullptr, nullptr, nullptr, nullptr,
                                                  Y, Z);

  // ---- MLP head ----
  gemm_kernel<1, 128, 2><<<gMT, 256, 0, stream>>>(Z, nullptr, nullptr, Wbm1, bm1,
                                                  nullptr, nullptr, nullptr, nullptr,
                                                  nullptr, H);
  gemm_kernel<1, 64, 3><<<gMT, 128, 0, stream>>>(H, nullptr, nullptr, Wbm2, bm2,
                                                 nullptr, nullptr, nullptr, nullptr,
                                                 nullptr, out);
}
